// MoleculeModel_53334903882077
// MI455X (gfx1250) — compile-verified, matrix operand order checked by reading, not run on hardware
//
#include <hip/hip_runtime.h>
#include <hip/hip_bf16.h>

typedef __attribute__((ext_vector_type(16))) _Float16     v16h;
typedef __attribute__((ext_vector_type(8)))  float        v8f;
typedef __attribute__((ext_vector_type(4)))  unsigned int u32x4;
typedef __attribute__((ext_vector_type(4)))  unsigned int tdm_u32x4;
typedef __attribute__((ext_vector_type(8)))  int          tdm_i32x8;
typedef __attribute__((ext_vector_type(4)))  int          tdm_i32x4;

#define N_ATOMS 500000
#define N_MOLS  25000
#define HID     300
#define KP      320   // K padded to multiple of 32 (WMMA f16 K=32)
#define NP      304   // hidden-layer width padded to multiple of 16
#define NT      (NP / 16)  // 19 N-tiles
#define TM      256   // atoms per workgroup
#define THREADS 256   // 8 waves (wave32)

#if __has_builtin(__builtin_amdgcn_tensor_load_to_lds) && __has_builtin(__builtin_amdgcn_s_wait_tensorcnt)
#define USE_TDM 1
#else
#define USE_TDM 0
#endif

// ---------------- kernel 0: convert+transpose weights to f16 [NP x KP] ----------------
// W1t[n][k] = W1[k][n]; zero-padded so WMMA tail tiles contribute exact zeros.
__global__ void cvt_weights_k(const float* __restrict__ W1,
                              const float* __restrict__ V1,
                              _Float16* __restrict__ W1t,
                              _Float16* __restrict__ V1t) {
  int idx = blockIdx.x * blockDim.x + threadIdx.x;
  if (idx >= NP * KP) return;
  int n = idx / KP, k = idx - n * KP;
  float w = 0.f, v = 0.f;
  if (n < HID && k < HID) { w = W1[k * HID + n]; v = V1[k * HID + n]; }
  W1t[idx] = (_Float16)w;
  V1t[idx] = (_Float16)v;
}

#if USE_TDM
// Issue a TDM 2D tile load: 16 rows x KP halves (row stride KP), global f16 -> LDS.
// D# packing per CDNA5 ISA (groups 0/1; groups 2/3 + extra group zero
// for a 2D tile). This toolchain exposes the 6-arg builtin:
//   (uint32x4 g0, int32x8 g1, int32x4 g2, int32x4 g3, int32x8 gx, i32 cpol)
__device__ __forceinline__ void tdm_load_slab(const _Float16* gsrc, unsigned int lds_off) {
  unsigned long long ga = (unsigned long long)(uintptr_t)gsrc;
  tdm_u32x4 g0;
  g0[0] = 1u;                                         // count=1 (valid), user mode
  g0[1] = lds_off;                                    // lds_addr (bytes)
  g0[2] = (unsigned int)(ga & 0xFFFFFFFFu);           // global_addr[31:0]
  g0[3] = (unsigned int)((ga >> 32) & 0x1FFFFFFu)     // global_addr[56:32]
        | (2u << 30);                                 // type=2 ("image")
  tdm_i32x8 g1;
  g1[0] = (int)(1u << 16);                            // wg_mask=0, data_size=1 (2 bytes)
  g1[1] = (int)((unsigned int)KP << 16);              // tensor_dim0[15:0] (=KP)
  g1[2] = (int)(16u << 16);                           // tensor_dim0 hi=0 | tensor_dim1[15:0]=16
  g1[3] = (int)((unsigned int)KP << 16);              // tensor_dim1 hi=0 | tile_dim0=KP
  g1[4] = (int)16u;                                   // tile_dim1=16, tile_dim2=0
  g1[5] = (int)KP;                                    // tensor_dim0_stride[31:0] (data_size units)
  g1[6] = 0;                                          // stride hi | tensor_dim1_stride lo
  g1[7] = 0;
  tdm_i32x4 z4 = {0, 0, 0, 0};
  tdm_i32x8 z8 = {0, 0, 0, 0, 0, 0, 0, 0};
  __builtin_amdgcn_tensor_load_to_lds(g0, g1, z4, z4, z8, 0);
}
#endif

// ---------------- kernel 1: fused dual-FFN (GEMM + bias + ReLU + 300->1 readout) ------
__global__ __launch_bounds__(THREADS) void fused_ffn_k(
    const float* __restrict__ hidden,
    const _Float16* __restrict__ W1t, const _Float16* __restrict__ V1t,
    const float* __restrict__ b1, const float* __restrict__ W2, const float* __restrict__ b2,
    const float* __restrict__ vb1, const float* __restrict__ V2, const float* __restrict__ vb2,
    float* __restrict__ out_o, float* __restrict__ out_w) {
  extern __shared__ _Float16 smem[];
  _Float16* shA = smem;                 // TM x KP  (hidden tile, f16)
  _Float16* shB = shA + TM * KP;        // 2 bufs x 2 mats x (16 x KP)

  const int tid  = threadIdx.x;
  const int wave = tid >> 5;
  const int lane = tid & 31;
  const int hf   = lane >> 4;            // lane half (K-offset selector)
  const int lr   = lane & 15;
  const long a0  = (long)blockIdx.x * TM;

#if USE_TDM
  // kick off slab fetch for N-tile 0 (buffer 0) before the big hidden staging loop
  if (wave == 0) {
    const unsigned int slab_bytes = 16 * KP * sizeof(_Float16);
    unsigned int base = (unsigned int)(uintptr_t)shB;   // LDS offset = low 32 bits (aperture rule)
    tdm_load_slab(W1t, base);
    tdm_load_slab(V1t, base + slab_bytes);
  }
#endif

  // ---- stage hidden tile to LDS as f16 (zero-pad K 300..319 and rows past N) ----
  for (int idx = tid; idx < TM * (KP / 2); idx += THREADS) {
    int row = idx / (KP / 2);
    int k   = (idx - row * (KP / 2)) * 2;
    float x0 = 0.f, x1 = 0.f;
    long atom = a0 + row;
    if (atom < N_ATOMS && k < HID) {        // HID even -> pair fully in-bounds
      const float* src = hidden + atom * (long)HID + k;
      x0 = src[0]; x1 = src[1];
    }
    shA[row * KP + k]     = (_Float16)x0;
    shA[row * KP + k + 1] = (_Float16)x1;
  }

  // per-lane layer-2 accumulators: [subtile][row-in-D]
  float accO0[8] = {}, accO1[8] = {}, accW0[8] = {}, accW1[8] = {};

  const _Float16* pa0 = shA + (wave * 32 + lr) * KP;        // subtile 0: rows wave*32..+15
  const _Float16* pa1 = shA + (wave * 32 + 16 + lr) * KP;   // subtile 1
  const int hk = hf * 8;   // 16-bit A layout: lane half -> K 0..7 / 8..15

  for (int j = 0; j < NT; ++j) {
    const int cur = j & 1;
    __syncthreads();   // staging done (j==0) / previous compute done (buffer reuse safe)

#if USE_TDM
    if (wave == 0) {
      if (j + 1 < NT) {
        // prefetch next N-tile's slabs into the other buffer, overlapped with compute
        const unsigned int slab_bytes = 16 * KP * sizeof(_Float16);
        unsigned int nbase = (unsigned int)(uintptr_t)(shB + ((j + 1) & 1) * 2 * 16 * KP);
        tdm_load_slab(W1t + (size_t)((j + 1) * 16) * KP, nbase);
        tdm_load_slab(V1t + (size_t)((j + 1) * 16) * KP, nbase + slab_bytes);
        __builtin_amdgcn_s_wait_tensorcnt(2);   // current buffer's 2 loads complete (in-order)
      } else {
        __builtin_amdgcn_s_wait_tensorcnt(0);
      }
    }
#else
    // fallback: cooperative copy of this tile's slabs
    for (int idx = tid; idx < 16 * (KP / 8); idx += THREADS) {   // KP halves = KP/8 x 16B
      int r = idx / (KP / 8), q = idx - r * (KP / 8);
      ((u32x4*)(shB + (cur * 2 + 0) * 16 * KP))[r * (KP / 8) + q] =
          ((const u32x4*)(W1t + (size_t)(j * 16 + r) * KP))[q];
      ((u32x4*)(shB + (cur * 2 + 1) * 16 * KP))[r * (KP / 8) + q] =
          ((const u32x4*)(V1t + (size_t)(j * 16 + r) * KP))[q];
    }
#endif
    __syncthreads();   // current slabs visible to all waves

    const _Float16* pb0 = shB + (cur * 2 + 0) * 16 * KP + lr * KP;
    const _Float16* pb1 = shB + (cur * 2 + 1) * 16 * KP + lr * KP;

    v8f d00 = {}, d01 = {}, d10 = {}, d11 = {};   // [ffn][subtile] accumulators (f32)
    for (int ks = 0; ks < KP / 32; ++ks) {
      const int kb = ks * 32 + hk;
      union { v16h v; u32x4 q[2]; } A0, A1, B0, B1;
      // ISA 16-bit fragment: halves 0..7 = K kb..kb+7, halves 8..15 = K kb+16..kb+23
      A0.q[0] = *(const u32x4*)(pa0 + kb); A0.q[1] = *(const u32x4*)(pa0 + kb + 16);
      A1.q[0] = *(const u32x4*)(pa1 + kb); A1.q[1] = *(const u32x4*)(pa1 + kb + 16);
      B0.q[0] = *(const u32x4*)(pb0 + kb); B0.q[1] = *(const u32x4*)(pb0 + kb + 16);
      B1.q[0] = *(const u32x4*)(pb1 + kb); B1.q[1] = *(const u32x4*)(pb1 + kb + 16);
      d00 = __builtin_amdgcn_wmma_f32_16x16x32_f16(false, A0.v, false, B0.v, (short)0, d00, false, false);
      d01 = __builtin_amdgcn_wmma_f32_16x16x32_f16(false, A1.v, false, B0.v, (short)0, d01, false, false);
      d10 = __builtin_amdgcn_wmma_f32_16x16x32_f16(false, A0.v, false, B1.v, (short)0, d10, false, false);
      d11 = __builtin_amdgcn_wmma_f32_16x16x32_f16(false, A1.v, false, B1.v, (short)0, d11, false, false);
    }

    // ---- fuse bias + ReLU + layer-2 dot product (column n = j*16 + lr per lane) ----
    const int n = j * 16 + lr;
    float b1n = 0.f, w2n = 0.f, vb1n = 0.f, v2n = 0.f;
    if (n < HID) { b1n = b1[n]; w2n = W2[n]; vb1n = vb1[n]; v2n = V2[n]; }
#pragma unroll
    for (int r = 0; r < 8; ++r) {
      float h;
      h = d00[r] + b1n;  accO0[r] += fmaxf(h, 0.f) * w2n;
      h = d01[r] + b1n;  accO1[r] += fmaxf(h, 0.f) * w2n;
      h = d10[r] + vb1n; accW0[r] += fmaxf(h, 0.f) * v2n;
      h = d11[r] + vb1n; accW1[r] += fmaxf(h, 0.f) * v2n;
    }
  }

  // ---- butterfly reduction over the 16 columns held per lane half, then store ----
  const float bb2 = b2[0], vbb2 = vb2[0];
#pragma unroll
  for (int r = 0; r < 8; ++r) {
    float so0 = accO0[r], so1 = accO1[r], sw0 = accW0[r], sw1 = accW1[r];
#pragma unroll
    for (int m = 1; m <= 8; m <<= 1) {     // stays within each 16-lane half
      so0 += __shfl_xor(so0, m, 32);
      so1 += __shfl_xor(so1, m, 32);
      sw0 += __shfl_xor(sw0, m, 32);
      sw1 += __shfl_xor(sw1, m, 32);
    }
    if (lr == 0) {
      // D layout: VGPR r, lane half hf -> row r + 8*hf of the 16-row subtile
      long m0 = a0 + wave * 32 + hf * 8 + r;
      long m1 = a0 + wave * 32 + 16 + hf * 8 + r;
      if (m0 < N_ATOMS) { out_o[m0] = so0 + bb2; out_w[m0] = sw0 + vbb2; }
      if (m1 < N_ATOMS) { out_o[m1] = so1 + bb2; out_w[m1] = sw1 + vbb2; }
    }
  }
}

// ---------------- small epilogue kernels ----------------
__global__ void zero_f32_k(float* p, int n) {
  int i = blockIdx.x * blockDim.x + threadIdx.x;
  if (i < n) p[i] = 0.f;
}

__global__ void seg_sum_k(const float* __restrict__ out_o, const float* __restrict__ out_w,
                          const int* __restrict__ seg,
                          float* __restrict__ sum_o, float* __restrict__ sum_w) {
  int i = blockIdx.x * blockDim.x + threadIdx.x;
  if (i < N_ATOMS) {
    int s = seg[i];
    atomicAdd(&sum_o[s], out_o[i]);
    atomicAdd(&sum_w[s], out_w[i]);
  }
}

__global__ void corr_k(const float* __restrict__ sum_o, const float* __restrict__ sum_w,
                       float* __restrict__ corr) {
  int i = blockIdx.x * blockDim.x + threadIdx.x;
  if (i < N_MOLS) {
    float sw = sum_w[i];
    if (sw == 0.f) sw = 1.f;
    corr[i] = (0.0f - sum_o[i]) / sw;   // CONSTRAINT == 0
  }
}

__global__ void final_k(const float* __restrict__ out_o, const float* __restrict__ out_w,
                        const int* __restrict__ seg, const float* __restrict__ corr,
                        float* __restrict__ out) {
  int i = blockIdx.x * blockDim.x + threadIdx.x;
  if (i < N_ATOMS) out[i] = out_o[i] + out_w[i] * corr[seg[i]];
}

// ---------------- launch ----------------
extern "C" void kernel_launch(void* const* d_in, const int* in_sizes, int n_in,
                              void* d_out, int out_size, void* d_ws, size_t ws_size,
                              hipStream_t stream) {
  (void)in_sizes; (void)n_in; (void)out_size; (void)ws_size;
  const float* hidden = (const float*)d_in[0];
  const int*   seg    = (const int*)  d_in[1];
  const float* W1     = (const float*)d_in[2];
  const float* b1     = (const float*)d_in[3];
  const float* W2     = (const float*)d_in[4];
  const float* b2     = (const float*)d_in[5];
  const float* V1     = (const float*)d_in[6];
  const float* vb1    = (const float*)d_in[7];
  const float* V2     = (const float*)d_in[8];
  const float* vb2    = (const float*)d_in[9];
  float* out = (float*)d_out;

  char* ws = (char*)d_ws;
  size_t off = 0;
  _Float16* W1t = (_Float16*)(ws + off); off += (size_t)NP * KP * sizeof(_Float16);
  _Float16* V1t = (_Float16*)(ws + off); off += (size_t)NP * KP * sizeof(_Float16);
  float* out_o = (float*)(ws + off); off += (size_t)N_ATOMS * sizeof(float);
  float* out_w = (float*)(ws + off); off += (size_t)N_ATOMS * sizeof(float);
  float* sum_o = (float*)(ws + off); off += (size_t)N_MOLS * sizeof(float);
  float* sum_w = (float*)(ws + off); off += (size_t)N_MOLS * sizeof(float);
  float* corr  = (float*)(ws + off); off += (size_t)N_MOLS * sizeof(float);

  cvt_weights_k<<<(NP * KP + 255) / 256, 256, 0, stream>>>(W1, V1, W1t, V1t);
  zero_f32_k<<<(2 * N_MOLS + 255) / 256, 256, 0, stream>>>(sum_o, 2 * N_MOLS); // sum_o+sum_w contiguous

  const int nblocks = (N_ATOMS + TM - 1) / TM;                                  // 1954
  const size_t smem = (size_t)(TM * KP + 4 * 16 * KP) * sizeof(_Float16);       // 204800 B < 320 KB/WGP
  fused_ffn_k<<<nblocks, THREADS, smem, stream>>>(hidden, W1t, V1t,
                                                  b1, W2, b2, vb1, V2, vb2,
                                                  out_o, out_w);

  seg_sum_k<<<(N_ATOMS + 255) / 256, 256, 0, stream>>>(out_o, out_w, seg, sum_o, sum_w);
  corr_k<<<(N_MOLS + 255) / 256, 256, 0, stream>>>(sum_o, sum_w, corr);
  final_k<<<(N_ATOMS + 255) / 256, 256, 0, stream>>>(out_o, out_w, seg, corr, out);
}
